// ScaledDotProductAttention_48679159333219
// MI455X (gfx1250) — matrix operand order repaired and checked by reading, not compiled, not run on hardware
//
#include <hip/hip_runtime.h>

// SDPA fwd: context = softmax(QK^T/sqrt(D) + mask) V, also outputs attn_prob.
// B=2,H=16,S=2048,D=64, fp32 in/out. WMMA f16 math, f32 accumulate.
#define B_ 2
#define H_ 16
#define S_ 2048
#define D_ 64

typedef __attribute__((ext_vector_type(16))) _Float16 v16h;
typedef __attribute__((ext_vector_type(8)))  float    v8f;

#define WMMA_F16(Afrag, Bfrag, Cfrag) \
  __builtin_amdgcn_wmma_f32_16x16x32_f16(false, (Afrag), false, (Bfrag), (short)0, (Cfrag), false, false)

// Pack 4x float4 (f32) -> 16 halves in order.
__device__ __forceinline__ v16h cvt16(const float4 a0, const float4 a1,
                                      const float4 b0, const float4 b1) {
  v16h r;
  r[0]=(_Float16)a0.x; r[1]=(_Float16)a0.y; r[2]=(_Float16)a0.z; r[3]=(_Float16)a0.w;
  r[4]=(_Float16)a1.x; r[5]=(_Float16)a1.y; r[6]=(_Float16)a1.z; r[7]=(_Float16)a1.w;
  r[8]=(_Float16)b0.x; r[9]=(_Float16)b0.y; r[10]=(_Float16)b0.z; r[11]=(_Float16)b0.w;
  r[12]=(_Float16)b1.x; r[13]=(_Float16)b1.y; r[14]=(_Float16)b1.z; r[15]=(_Float16)b1.w;
  return r;
}

// A-fragment (16x32 f16): lane holds row M=lane&15; a[i]=K(b8+i), i=0..15, b8=(lane&16)?16:0.
// rowp points at the 32-float (one K-chunk) slice of a row-major f32 row.
__device__ __forceinline__ v16h load_afrag(const float* __restrict__ rowp, int lane) {
  const int b8 = (lane & 16) ? 16 : 0;
  const float4 x0 = *reinterpret_cast<const float4*>(rowp + b8);
  const float4 x1 = *reinterpret_cast<const float4*>(rowp + b8 + 4);
  const float4 y0 = *reinterpret_cast<const float4*>(rowp + b8 + 8);
  const float4 y1 = *reinterpret_cast<const float4*>(rowp + b8 + 12);
  return cvt16(x0, x1, y0, y1);
}

// B-fragment (32x16 f16): lane holds column n=lane&15; b[j]=K(off+j), off=(lane&16)?16:0.
// p points at 16 contiguous f32 (caller applied off).
__device__ __forceinline__ v16h load_bfrag16(const float* __restrict__ p) {
  const float4 x0 = *reinterpret_cast<const float4*>(p);
  const float4 x1 = *reinterpret_cast<const float4*>(p + 4);
  const float4 x2 = *reinterpret_cast<const float4*>(p + 8);
  const float4 x3 = *reinterpret_cast<const float4*>(p + 12);
  return cvt16(x0, x1, x2, x3);
}

// probs for one 16-key score subtile held in C-layout: p[i] = exp(mask?c*scale:-1e9 - mt) * inv_l
__device__ __forceinline__ void probs8(const v8f c, const int* __restrict__ mp,
                                       float mt, float inv_l, float* p) {
  const int4 m0 = *reinterpret_cast<const int4*>(mp);
  const int4 m1 = *reinterpret_cast<const int4*>(mp + 4);
  const int mk[8] = {m0.x, m0.y, m0.z, m0.w, m1.x, m1.y, m1.z, m1.w};
#pragma unroll
  for (int i = 0; i < 8; ++i) {
    const float s = mk[i] ? c[i] * 0.125f : -1e9f;
    p[i] = __expf(s - mt) * inv_l;   // exp(-huge) underflows to 0 for masked
  }
}

__global__ __launch_bounds__(128)
void ScaledDotProductAttention_48679159333219_kernel(
    const float* __restrict__ Q, const float* __restrict__ K,
    const float* __restrict__ V, const int* __restrict__ mask,
    float* __restrict__ ctx_out, float* __restrict__ prob_out) {
  const int lane  = threadIdx.x & 31;
  const int wave  = threadIdx.x >> 5;
  const int qt    = blockIdx.x % (S_ / 64);
  const int bh    = blockIdx.x / (S_ / 64);
  const int bidx  = bh / H_;

  const int n16   = lane & 15;           // fixed q-column for scores^T
  const int base8 = (lane & 16) ? 8 : 0; // key sub-range per lane half (C/A frags)
  const int off16 = (lane & 16) ? 16 : 0;// K-range per lane half (B frags)

  const int q0   = qt * 64 + wave * 16;  // first q row of this wave's tile
  const int qrow = q0 + n16;             // this lane's q row

  const float* Qbh = Q + (size_t)bh * S_ * D_;
  const float* Kbh = K + (size_t)bh * S_ * D_;
  const float* Vbh = V + (size_t)bh * S_ * D_;
  const int* maskrow = mask + ((size_t)bidx * S_ + qrow) * S_;

  // Q^T B-fragments (held for the whole kernel): Q[qrow][off16 + 32*chunk + 0..15]
  const float* qp = Qbh + (size_t)qrow * D_ + off16;
  const v16h qb0 = load_bfrag16(qp);
  const v16h qb1 = load_bfrag16(qp + 32);

  // ---------------- pass 1: row max + exp-sum (online, in-lane) ----------------
  float m = -1e30f, l = 0.f;
#pragma unroll 2
  for (int kb = 0; kb < S_; kb += 16) {
    const float* krow = Kbh + (size_t)(kb + n16) * D_;
    if (kb + 16 < S_) __builtin_prefetch(krow + 16 * D_, 0, 0); // global_prefetch_b8
    v8f c = {};
    c = WMMA_F16(load_afrag(krow,      lane), qb0, c);
    c = WMMA_F16(load_afrag(krow + 32, lane), qb1, c);
    const int4 m0 = *reinterpret_cast<const int4*>(maskrow + kb + base8);
    const int4 m1 = *reinterpret_cast<const int4*>(maskrow + kb + base8 + 4);
    const int mk[8] = {m0.x, m0.y, m0.z, m0.w, m1.x, m1.y, m1.z, m1.w};
    float s[8];
#pragma unroll
    for (int i = 0; i < 8; ++i) s[i] = mk[i] ? c[i] * 0.125f : -1e9f;
    float nm = m;
#pragma unroll
    for (int i = 0; i < 8; ++i) nm = fmaxf(nm, s[i]);
    float acc = 0.f;
#pragma unroll
    for (int i = 0; i < 8; ++i) acc += __expf(s[i] - nm);
    l = l * __expf(m - nm) + acc;
    m = nm;
  }
  // merge the two lane halves (same q row, complementary key subsets)
  const float mo = __shfl_xor(m, 16, 32);
  const float lo = __shfl_xor(l, 16, 32);
  const float mt = fmaxf(m, mo);
  const float lt = l * __expf(m - mt) + lo * __expf(mo - mt);
  const float inv_l = 1.0f / lt;

  // ---------------- pass 2: probs out + context = P·V ----------------
  v8f ctxA[4] = {v8f{}, v8f{}, v8f{}, v8f{}};   // D split into 4 n-tiles of 16
  float* probrow = prob_out + ((size_t)bh * S_ + qrow) * S_;

#pragma unroll 1
  for (int kb = 0; kb < S_; kb += 32) {
    const float* krow1 = Kbh + (size_t)(kb + n16) * D_;
    const float* krow2 = krow1 + 16 * D_;
    if (kb + 32 < S_) __builtin_prefetch(krow1 + 32 * D_, 0, 0);

    v8f c1 = {}, c2 = {};
    c1 = WMMA_F16(load_afrag(krow1,      lane), qb0, c1);
    c1 = WMMA_F16(load_afrag(krow1 + 32, lane), qb1, c1);
    c2 = WMMA_F16(load_afrag(krow2,      lane), qb0, c2);
    c2 = WMMA_F16(load_afrag(krow2 + 32, lane), qb1, c2);

    float p1[8], p2[8];
    probs8(c1, maskrow + kb + base8,      mt, inv_l, p1);
    probs8(c2, maskrow + kb + 16 + base8, mt, inv_l, p2);

    // attn_prob: 8 contiguous f32 per half-lane per subtile
    *reinterpret_cast<float4*>(probrow + kb + base8)          = make_float4(p1[0], p1[1], p1[2], p1[3]);
    *reinterpret_cast<float4*>(probrow + kb + base8 + 4)      = make_float4(p1[4], p1[5], p1[6], p1[7]);
    *reinterpret_cast<float4*>(probrow + kb + 16 + base8)     = make_float4(p2[0], p2[1], p2[2], p2[3]);
    *reinterpret_cast<float4*>(probrow + kb + 16 + base8 + 4) = make_float4(p2[4], p2[5], p2[6], p2[7]);

    // Exp'ed C-frags of two 16-key subtiles == A-fragment (16x32) of P for this 32-key chunk.
    v16h pa;
#pragma unroll
    for (int i = 0; i < 8; ++i) { pa[i] = (_Float16)p1[i]; pa[8 + i] = (_Float16)p2[i]; }

    // V B-fragments: b[j] = V[kb + base8 + j + (j&8)][n16 + 16*dt]
    const float* vcol = Vbh + (size_t)(kb + base8) * D_ + n16;
#pragma unroll
    for (int dt = 0; dt < 4; ++dt) {
      v16h vb;
#pragma unroll
      for (int j = 0; j < 16; ++j) vb[j] = (_Float16)vcol[(size_t)(j + (j & 8)) * D_ + dt * 16];
      ctxA[dt] = WMMA_F16(pa, vb, ctxA[dt]);
    }
  }

  // context store: lane holds q = q0 + base8 + r (per VGPR r), dim = n16 + 16*dt
  float* crow = ctx_out + ((size_t)bh * S_ + q0 + base8) * D_ + n16;
#pragma unroll
  for (int r = 0; r < 8; ++r) {
    crow[(size_t)r * D_ +  0] = ctxA[0][r];
    crow[(size_t)r * D_ + 16] = ctxA[1][r];
    crow[(size_t)r * D_ + 32] = ctxA[2][r];
    crow[(size_t)r * D_ + 48] = ctxA[3][r];
  }
}

extern "C" void kernel_launch(void* const* d_in, const int* in_sizes, int n_in,
                              void* d_out, int out_size, void* d_ws, size_t ws_size,
                              hipStream_t stream) {
  (void)in_sizes; (void)n_in; (void)d_ws; (void)ws_size; (void)out_size;
  const float* Q   = (const float*)d_in[0];
  const float* K   = (const float*)d_in[1];
  const float* V   = (const float*)d_in[2];
  const int* mask  = (const int*)d_in[3];
  float* ctx  = (float*)d_out;                              // [B,H,S,D]
  float* prob = ctx + (size_t)B_ * H_ * S_ * D_;            // [B,H,S,S]
  dim3 grid(B_ * H_ * (S_ / 64));
  dim3 block(128);
  ScaledDotProductAttention_48679159333219_kernel<<<grid, block, 0, stream>>>(
      Q, K, V, mask, ctx, prob);
}
